// MERG_80874234183746
// MI455X (gfx1250) — matrix operand order repaired and checked by reading, not compiled, not run on hardware
//
#include <hip/hip_runtime.h>
#include <hip/hip_bf16.h>

typedef __attribute__((ext_vector_type(16))) __bf16 v16bf;
typedef __attribute__((ext_vector_type(8)))  float  v8f;

#define HD 32          // hidden dim
#define TILES 64       // 1024 / 16 output-column tiles of the big GEMM
#define EPW 16         // edges per wave
#define WPB 8          // waves per block
#define EPB (EPW*WPB)  // 128 edges per block

// Fragment image layout (bf16), built once in workspace, copied to LDS per block:
//   [0      , 32768) : W1^T A-fragments, ldsA[t][lane][i]    (64 KB)
//   [32768  , 33792) : W3   B-fragments, [nt][lane][i]       ( 2 KB)
//   [33792  , 35840) : W_edge B-fragments, [kh][nt][lane][i] ( 4 KB)
//   [35840  , 36864) : zero padding (keeps copy loop exact)
#define FRAG_ELEMS 36864

// ---------------------------------------------------------------------------
// Kernel 0: P2 = h @ W2 + b2  (N x 32), and zero the BN stats accumulators.
// ---------------------------------------------------------------------------
__global__ void gnn_p2_kernel(const float* __restrict__ h,
                              const float* __restrict__ W2,
                              const float* __restrict__ b2,
                              float* __restrict__ P2,
                              float* __restrict__ stats,
                              int N) {
    int gid = blockIdx.x * 256 + threadIdx.x;
    if (gid < 64) stats[gid] = 0.0f;          // sum[32], sumsq[32]
    if (gid >= N * HD) return;
    int n = gid >> 5;
    int c = gid & 31;
    const float* hr = h + (size_t)n * HD;
    float acc = b2[c];
#pragma unroll
    for (int j = 0; j < HD; ++j) acc += hr[j] * W2[j * HD + c];
    P2[gid] = acc;
}

// ---------------------------------------------------------------------------
// Kernel 0b: bb[n][kk] = sum_d b1[kk*32+d] * P2[n][d]   (dst-node bias term)
// ---------------------------------------------------------------------------
__global__ void gnn_bb_kernel(const float* __restrict__ P2,
                              const float* __restrict__ b1,
                              float* __restrict__ bb,
                              int N) {
    int gid = blockIdx.x * 256 + threadIdx.x;
    if (gid >= N * HD) return;
    int n  = gid >> 5;
    int kk = gid & 31;
    const float* pr = P2 + (size_t)n * HD;
    const float* br = b1 + kk * HD;
    float acc = 0.0f;
#pragma unroll
    for (int d = 0; d < HD; ++d) acc += br[d] * pr[d];
    bb[gid] = acc;
}

// ---------------------------------------------------------------------------
// Kernel 0c: one-shot bf16 fragment packing into workspace.
// ---------------------------------------------------------------------------
__global__ void gnn_pack_kernel(const float* __restrict__ W1,
                                const float* __restrict__ W3,
                                const float* __restrict__ W_edge,
                                __bf16* __restrict__ frag) {
    int gid = blockIdx.x * 256 + threadIdx.x;
    if (gid >= FRAG_ELEMS) return;
    float v = 0.0f;
    if (gid < 32768) {
        // A-matrix 16x32 bf16 layout: lane ln -> row m=ln&15;
        // elt i -> k = 16*(ln>>4) + i;  A[m][k] = W1[k][16t+m]
        int i  = gid & 15;
        int ln = (gid >> 4) & 31;
        int t  = gid >> 9;
        int k  = 16 * (ln >> 4) + i;
        int j  = t * 16 + (ln & 15);
        v = W1[k * 1024 + j];
    } else if (gid < 33792) {
        // B-matrix 32x16 bf16 layout: lane -> col n=ln&15; elt i -> k=16*(ln>>4)+i
        int idx = gid - 32768;
        int i  = idx & 15;
        int ln = (idx >> 4) & 31;
        int nt = idx >> 9;
        int k  = 16 * (ln >> 4) + i;
        int n  = nt * 16 + (ln & 15);
        v = W3[k * HD + n];
    } else if (gid < 35840) {
        int idx = gid - 33792;
        int i  = idx & 15;
        int ln = (idx >> 4) & 31;
        int nt = (idx >> 9) & 1;
        int kh = idx >> 10;
        int k  = kh * 32 + 16 * (ln >> 4) + i;
        int n  = nt * 16 + (ln & 15);
        v = W_edge[k * HD + n];
    }
    frag[gid] = (__bf16)v;
}

// ---------------------------------------------------------------------------
// Kernel 1: fused edge kernel (one wave32 per 16 edges, 70 bf16 WMMAs/wave).
// min 2 waves/EU guarantees the 8-wave workgroup fits on a WGP; no VGPR cap
// (a 256 cap provably causes scratch spills of WMMA D-fragments).
// ---------------------------------------------------------------------------
__global__ void __launch_bounds__(256, 2)
gnn_edge_kernel(const float* __restrict__ h,
                const int*   __restrict__ feat,
                const int*   __restrict__ src_idx,
                const int*   __restrict__ dst_idx,
                const float* __restrict__ emb_src,
                const float* __restrict__ emb_dst,
                const float* __restrict__ b_edge,
                const float* __restrict__ b3,
                const float* __restrict__ P2,
                const float* __restrict__ bb,
                const __bf16* __restrict__ frag,
                float* __restrict__ out,
                float* __restrict__ stats,
                int E) {
    __shared__ __attribute__((aligned(32))) __bf16 ldsF[FRAG_ELEMS]; // 72 KB
    __shared__ float sBuf[HD];
    __shared__ float sqBuf[HD];

    const int tid = threadIdx.x;
    if (tid < HD) { sBuf[tid] = 0.0f; sqBuf[tid] = 0.0f; }

    // coalesced 16B copy of the packed fragment image into LDS
    {
        const int4* g4 = (const int4*)frag;
        int4* l4 = (int4*)ldsF;
#pragma unroll
        for (int i = tid; i < FRAG_ELEMS / 8; i += 256) l4[i] = g4[i];
    }
    __syncthreads();

    const __bf16* ldsA  = ldsF;            // [64][32][16]
    const __bf16* ldsW3 = ldsF + 32768;    // [2][32][16]
    const __bf16* ldsWE = ldsF + 33792;    // [2][2][32][16]

    const int wave = tid >> 5;
    const int ln   = tid & 31;
    const int hi   = ln >> 4;          // lane half
    const int e    = ln & 15;          // edge slot within tile
    const int edge_base = (blockIdx.x * WPB + wave) * EPW;
    const bool active = (edge_base < E);   // uniform per wave (E % 128 == 0)

    if (active) {
        const int eidx = edge_base + e;
        const int sn = src_idx[eidx];
        const int dn = dst_idx[eidx];
        const int fs = feat[sn];
        const int fd = feat[dn];

        // B fragment: h[src]^T (32 x 16edges). lane col = e, k = 16*hi + i
        v16bf bfrag;
        {
            const float* hr = h + (size_t)sn * HD + 16 * hi;
#pragma unroll
            for (int i = 0; i < 16; ++i) bfrag[i] = (__bf16)hr[i];
        }
        // P2[dst] slices this lane needs: d = {8hi..8hi+7} u {16+8hi..16+8hi+7}
        float p2r[16];
        {
            const float* pr = P2 + (size_t)dn * HD;
#pragma unroll
            for (int i = 0; i < 8; ++i) {
                p2r[i]     = pr[8 * hi + i];
                p2r[8 + i] = pr[16 + 8 * hi + i];
            }
        }
        // Embedding A-fragments (row m = e, A-layout k selection)
        v16bf asrc, adst;
        {
            const float* es = emb_src + (size_t)fs * HD;
            const float* ed = emb_dst + (size_t)fd * HD;
#pragma unroll
            for (int i = 0; i < 16; ++i) {
                int k = 16 * hi + i;
                asrc[i] = (__bf16)es[k];
                adst[i] = (__bf16)ed[k];
            }
        }

        // accumulators pre-seeded with 0.5*bb[dst][kk]; the lane^16 merge
        // below sums the two halves back to exactly bb[dst][kk].
        float accv[32];
        {
            const float4* bp = (const float4*)(bb + (size_t)dn * HD);
#pragma unroll
            for (int q = 0; q < 8; ++q) {
                float4 t = bp[q];
                accv[4 * q + 0] = 0.5f * t.x;
                accv[4 * q + 1] = 0.5f * t.y;
                accv[4 * q + 2] = 0.5f * t.z;
                accv[4 * q + 3] = 0.5f * t.w;
            }
        }

        const v8f czero = {0.f, 0.f, 0.f, 0.f, 0.f, 0.f, 0.f, 0.f};

        // ---- main loop: 64 WMMAs computing G^T tiles, contracted with P2 ----
#pragma unroll
        for (int kk = 0; kk < 32; ++kk) {
            float a = accv[kk];
#pragma unroll
            for (int p = 0; p < 2; ++p) {
                const int t = 2 * kk + p;
                v16bf afrag = *(const v16bf*)&ldsA[t * 512 + ln * 16];
                v8f d = __builtin_amdgcn_wmma_f32_16x16x32_bf16(
                    false, afrag, false, bfrag, (short)0, czero, false, false);
#pragma unroll
                for (int r = 0; r < 8; ++r) a += d[r] * p2r[p * 8 + r];
            }
            accv[kk] = a;
        }
        // merge partner-lane partial (lane^16 holds complementary d range)
#pragma unroll
        for (int kk = 0; kk < 32; ++kk)
            accv[kk] += __shfl_xor(accv[kk], 16, 32);

        // A-fragment of lr (row m = e, k selected per A-layout)
        v16bf alr;
#pragma unroll
        for (int i = 0; i < 16; ++i) {
            int k = 16 * hi + i;
            alr[i] = (__bf16)accv[k];
        }

        // ---- tail: lr@W3 + src@We[0:32] + dst@We[32:64] + (b3+b_edge) ----
#pragma unroll
        for (int nt = 0; nt < 2; ++nt) {
            const int c = nt * 16 + e;
            const float bias = b_edge[c] + b3[c];
            v8f cf;
#pragma unroll
            for (int r = 0; r < 8; ++r) cf[r] = bias;
            v16bf bw3 = *(const v16bf*)&ldsW3[nt * 512 + ln * 16];
            v16bf be0 = *(const v16bf*)&ldsWE[(0 * 2 + nt) * 512 + ln * 16];
            v16bf be1 = *(const v16bf*)&ldsWE[(1 * 2 + nt) * 512 + ln * 16];
            v8f d;
            d = __builtin_amdgcn_wmma_f32_16x16x32_bf16(false, alr,  false, bw3, (short)0, cf, false, false);
            d = __builtin_amdgcn_wmma_f32_16x16x32_bf16(false, asrc, false, be0, (short)0, d,  false, false);
            d = __builtin_amdgcn_wmma_f32_16x16x32_bf16(false, adst, false, be1, (short)0, d,  false, false);

            float s1 = 0.0f, s2 = 0.0f;
#pragma unroll
            for (int r = 0; r < 8; ++r) {
                const float v = d[r];
                const int m = r + 8 * hi;                    // edge within tile
                out[(size_t)(edge_base + m) * HD + c] = v;   // raw (pre-BN)
                s1 += v;
                s2 += v * v;
            }
            atomicAdd(&sBuf[c], s1);     // ds_add_f32
            atomicAdd(&sqBuf[c], s2);
        }
    }

    __syncthreads();
    if (tid < HD) {
        atomicAdd(&stats[tid], sBuf[tid]);
        atomicAdd(&stats[HD + tid], sqBuf[tid]);
    }
}

// ---------------------------------------------------------------------------
// Kernel 2: BatchNorm (batch stats, biased var) + ReLU, in place, float4.
// ---------------------------------------------------------------------------
__global__ void gnn_bn_relu_kernel(float* __restrict__ out,
                                   const float* __restrict__ stats,
                                   const float* __restrict__ gamma,
                                   const float* __restrict__ beta,
                                   int E) {
    __shared__ float sc[HD];
    __shared__ float sh[HD];
    const int tid = threadIdx.x;
    if (tid < HD) {
        const float invE = 1.0f / (float)E;
        float m = stats[tid] * invE;
        float v = stats[HD + tid] * invE - m * m;
        float g = gamma[tid] * rsqrtf(v + 1e-5f);
        sc[tid] = g;
        sh[tid] = beta[tid] - m * g;
    }
    __syncthreads();
    int gid = blockIdx.x * blockDim.x + tid;
    int total4 = E * (HD / 4);
    if (gid < total4) {
        float4 v = ((const float4*)out)[gid];
        int c0 = (gid * 4) & (HD - 1);
        v.x = fmaxf(fmaf(v.x, sc[c0 + 0], sh[c0 + 0]), 0.0f);
        v.y = fmaxf(fmaf(v.y, sc[c0 + 1], sh[c0 + 1]), 0.0f);
        v.z = fmaxf(fmaf(v.z, sc[c0 + 2], sh[c0 + 2]), 0.0f);
        v.w = fmaxf(fmaf(v.w, sc[c0 + 3], sh[c0 + 3]), 0.0f);
        ((float4*)out)[gid] = v;
    }
}

// ---------------------------------------------------------------------------
extern "C" void kernel_launch(void* const* d_in, const int* in_sizes, int n_in,
                              void* d_out, int out_size, void* d_ws, size_t ws_size,
                              hipStream_t stream) {
    const float* h       = (const float*)d_in[0];
    // d_in[1] = e, unused by the forward pass
    const int*   feat    = (const int*)  d_in[2];
    const int*   src_idx = (const int*)  d_in[3];
    const int*   dst_idx = (const int*)  d_in[4];
    const float* emb_src = (const float*)d_in[5];
    const float* emb_dst = (const float*)d_in[6];
    const float* W_edge  = (const float*)d_in[7];
    const float* b_edge  = (const float*)d_in[8];
    const float* W1      = (const float*)d_in[9];
    const float* b1      = (const float*)d_in[10];
    const float* W2      = (const float*)d_in[11];
    const float* b2      = (const float*)d_in[12];
    const float* W3      = (const float*)d_in[13];
    const float* b3      = (const float*)d_in[14];
    const float* gamma   = (const float*)d_in[15];
    const float* beta    = (const float*)d_in[16];

    const int N = in_sizes[0] / HD;   // 40000
    const int E = in_sizes[3];        // 320000

    float*  P2    = (float*)d_ws;                 // N*32 floats
    float*  bb    = P2 + (size_t)N * HD;          // N*32 floats
    float*  stats = bb + (size_t)N * HD;          // 64 floats
    __bf16* frag  = (__bf16*)(stats + 64);        // FRAG_ELEMS bf16 (16B-aligned)
    float*  out   = (float*)d_out;                // E*32 floats

    // Stage 0: P2 = h@W2 + b2, zero BN stats; pack bf16 fragments; bb term
    {
        int blocks = (N * HD + 255) / 256;
        gnn_p2_kernel<<<blocks, 256, 0, stream>>>(h, W2, b2, P2, stats, N);
        gnn_pack_kernel<<<(FRAG_ELEMS + 255) / 256, 256, 0, stream>>>(W1, W3, W_edge, frag);
        gnn_bb_kernel<<<blocks, 256, 0, stream>>>(P2, b1, bb, N);
    }
    // Stage 1: fused edge compute (WMMA) + raw out + stat accumulation
    {
        int blocks = (E + EPB - 1) / EPB;
        gnn_edge_kernel<<<blocks, 256, 0, stream>>>(
            h, feat, src_idx, dst_idx, emb_src, emb_dst,
            b_edge, b3, P2, bb, frag, out, stats, E);
    }
    // Stage 2: BatchNorm + ReLU in place
    {
        int blocks = (E * (HD / 4) + 255) / 256;
        gnn_bn_relu_kernel<<<blocks, 256, 0, stream>>>(out, stats, gamma, beta, E);
    }
}
